// CrossAttn_17557826306465
// MI455X (gfx1250) — matrix operand order repaired and checked by reading, not compiled, not run on hardware
//
#include <hip/hip_runtime.h>

// ---------------------------------------------------------------------------
// CDNA5 (gfx1250, wave32) WMMA types
// ---------------------------------------------------------------------------
typedef __bf16  v16bf __attribute__((ext_vector_type(16)));
typedef __bf16  v8bf  __attribute__((ext_vector_type(8)));
typedef float   v8f   __attribute__((ext_vector_type(8)));
typedef unsigned int u32x4 __attribute__((ext_vector_type(4)));
typedef int          i32x4 __attribute__((ext_vector_type(4)));
typedef int          i32x8 __attribute__((ext_vector_type(8)));

union BF16x16 { v16bf v; v8bf h[2]; };

__device__ __forceinline__ v8f wmma_bf16(v16bf a, v16bf b, v8f c) {
  // D = A(16x32 bf16) * B(32x16 bf16) + C(16x16 f32)
  return __builtin_amdgcn_wmma_f32_16x16x32_bf16(
      /*neg_a=*/false, a, /*neg_b=*/false, b,
      /*c_mod=*/(short)0, c, /*reuse_a=*/false, /*reuse_b=*/false);
}

// A-fragment (16x32 bf16). rowk points at the lane's matrix row, already
// offset by the 32-wide K-chunk base. Layout: lane l (m = l&15,
// hf = l>>4): elems 0..15 -> K = 16*hf + e (matches load_b_frag)
__device__ __forceinline__ v16bf load_a_frag(const __bf16* rowk, int hf) {
  BF16x16 r;
  r.h[0] = *(const v8bf*)(rowk + 16 * hf);
  r.h[1] = *(const v8bf*)(rowk + 16 * hf + 8);
  return r.v;
}

// B-fragment (32x16 bf16). p points at 16 consecutive K values for this lane:
// lane l needs B[K = 16*(l>>4) + e][N = l&15], so source must be K-contiguous.
__device__ __forceinline__ v16bf load_b_frag(const __bf16* p) {
  BF16x16 r;
  r.h[0] = *(const v8bf*)(p);
  r.h[1] = *(const v8bf*)(p + 8);
  return r.v;
}

// ---------------------------------------------------------------------------
// Tensor Data Mover: async 2D tile DMA global -> LDS (CDNA5 TDM, ISA ch.8)
// Descriptor built per ISA 8.3/8.4; groups 2/3 zero (2D tile, dims unused).
// ---------------------------------------------------------------------------
#if defined(__has_builtin)
#  if __has_builtin(__builtin_amdgcn_tensor_load_to_lds)
#    define ATHENA_HAS_TDM 1
#  endif
#endif
#ifndef ATHENA_HAS_TDM
#  define ATHENA_HAS_TDM 0
#endif
#if defined(__has_include)
#  if __has_include(<hip/amd_detail/amd_gfx1250_TDM.h>)
#    define ATHENA_TDM_6ARG 1
#  endif
#endif
#ifndef ATHENA_TDM_6ARG
#  define ATHENA_TDM_6ARG 0
#endif

#if ATHENA_HAS_TDM
// All sizes/strides in bf16 elements (data_size code 1 == 2 bytes).
__device__ __forceinline__ void tdm_load_2d_bf16(
    const __bf16* gsrc, const __bf16* ldst,
    unsigned tile_d0, unsigned tile_d1,
    unsigned tensor_d0, unsigned tensor_d1,
    unsigned long long stride0_elems) {
  unsigned long long ga = (unsigned long long)(size_t)gsrc;
  unsigned la = (unsigned)(size_t)ldst;  // low 32 bits of LDS generic ptr
  u32x4 g0;
  g0.x = 1u;                                      // count=1, user mode
  g0.y = la;                                      // [63:32] lds_addr
  g0.z = (unsigned)(ga & 0xffffffffu);            // [95:64] gaddr lo
  g0.w = (unsigned)((ga >> 32) & 0x01ffffffu)     // [120:96] gaddr hi
         | (2u << 30);                            // [127:126] type=2 (image)
  i32x8 g1;
  g1[0] = (int)(1u << 16);                        // data_size=1 (2B), mask=0
  g1[1] = (int)((tensor_d0 & 0xffffu) << 16);     // [63:48] tdim0 lo
  g1[2] = (int)((tensor_d0 >> 16) | ((tensor_d1 & 0xffffu) << 16));
  g1[3] = (int)((tensor_d1 >> 16) | (tile_d0 << 16));  // [127:112] tile0
  g1[4] = (int)(tile_d1 & 0xffffu);               // [143:128] tile1, tile2=0
  g1[5] = (int)(stride0_elems & 0xffffffffull);   // [191:160] stride0 lo
  g1[6] = (int)((stride0_elems >> 32) & 0xffffull);  // stride0 hi, stride1=0
  g1[7] = 0;
  i32x4 z4 = {0, 0, 0, 0};
#  if ATHENA_TDM_6ARG
  i32x8 z8 = {0, 0, 0, 0, 0, 0, 0, 0};
  __builtin_amdgcn_tensor_load_to_lds(g0, g1, z4, z4, z8, 0);
#  else
  __builtin_amdgcn_tensor_load_to_lds(g0, g1, z4, z4, 0);
#  endif
}
#endif

// ---------------------------------------------------------------------------
// Kernel 0: f32 -> bf16 conversion
// ---------------------------------------------------------------------------
__global__ void cvt_f32_bf16(const float* __restrict__ in,
                             __bf16* __restrict__ out, int n) {
  int i = blockIdx.x * blockDim.x + threadIdx.x;
  int stride = gridDim.x * blockDim.x;
  for (; i < n; i += stride) out[i] = (__bf16)in[i];
}

// ---------------------------------------------------------------------------
// Kernel 1/2: C(bf16)[M,N] = A(bf16)[M,K] @ B(bf16)[K,N], f32 accumulation.
// 256 threads (8 waves), 128x128 tile, BK=32; each wave does 64x32 (8 WMMAs).
// A tile staged by TDM DMA (wave 0), B tile transposed manually by all waves.
// ---------------------------------------------------------------------------
__global__ __launch_bounds__(256) void gemm_bf16(
    const __bf16* __restrict__ A, const __bf16* __restrict__ Bw,
    __bf16* __restrict__ C, int M, int N, int K) {
  __shared__ alignas(16) __bf16 lds_a[128][32];   // [m][k]
  __shared__ alignas(16) __bf16 lds_bt[128][32];  // [n][k] (transposed)

  const int tid  = threadIdx.x;
  const int wave = tid >> 5, lane = tid & 31;
  const int hf   = lane >> 4, ln = lane & 15;
  const int rowBase = blockIdx.y * 128, colBase = blockIdx.x * 128;
  const int wr = wave & 1;   // 64-row group
  const int wc = wave >> 1;  // 32-col group

  v8f acc[4][2] = {};

  for (int k0 = 0; k0 < K; k0 += 32) {
#if ATHENA_HAS_TDM
    if (wave == 0)  // async DMA: A tile 128 rows x 32 cols
      tdm_load_2d_bf16(A + (size_t)rowBase * K + k0, &lds_a[0][0],
                       /*tile*/ 32, 128, /*tensor*/ (unsigned)K, (unsigned)M,
                       (unsigned long long)K);
#else
    {  // batched manual A staging: issue both loads, then store
      uint4 ta[2];
#pragma unroll
      for (int i = 0; i < 2; ++i) {
        int idx = tid + i * 256;
        int r = idx >> 2, c8 = (idx & 3) * 8;
        ta[i] = *(const uint4*)(A + (size_t)(rowBase + r) * K + k0 + c8);
      }
#pragma unroll
      for (int i = 0; i < 2; ++i) {
        int idx = tid + i * 256;
        int r = idx >> 2, c8 = (idx & 3) * 8;
        *(uint4*)&lds_a[r][c8] = ta[i];
      }
    }
#endif
    {  // batched B staging with transpose: 32x128 bf16 = 512 uint4
      uint4 tb[2];
#pragma unroll
      for (int i = 0; i < 2; ++i) {
        int idx = tid + i * 256;
        int kk = idx >> 4, n8 = (idx & 15) * 8;
        tb[i] = *(const uint4*)(Bw + (size_t)(k0 + kk) * N + colBase + n8);
      }
#pragma unroll
      for (int i = 0; i < 2; ++i) {
        int idx = tid + i * 256;
        int kk = idx >> 4, n8 = (idx & 15) * 8;
        const __bf16* s = (const __bf16*)&tb[i];
#pragma unroll
        for (int j = 0; j < 8; ++j) lds_bt[n8 + j][kk] = s[j];
      }
    }
    if (k0 + 32 < K)  // global_prefetch_b8 of next B tile
      __builtin_prefetch(Bw + (size_t)(k0 + 32 + (tid >> 4)) * N + colBase, 0, 0);
#if ATHENA_HAS_TDM
    if (wave == 0) __builtin_amdgcn_s_wait_tensorcnt(0);
#endif
    __syncthreads();

    v16bf af[4], bf[2];
#pragma unroll
    for (int ms = 0; ms < 4; ++ms)
      af[ms] = load_a_frag(&lds_a[wr * 64 + ms * 16 + ln][0], hf);
#pragma unroll
    for (int ns = 0; ns < 2; ++ns)
      bf[ns] = load_b_frag(&lds_bt[wc * 32 + ns * 16 + ln][16 * hf]);
#pragma unroll
    for (int ms = 0; ms < 4; ++ms)
#pragma unroll
      for (int ns = 0; ns < 2; ++ns)
        acc[ms][ns] = wmma_bf16(af[ms], bf[ns], acc[ms][ns]);
    __syncthreads();
  }
  // store: C/D layout -> M = e + 8*hf, N = ln
#pragma unroll
  for (int ms = 0; ms < 4; ++ms)
#pragma unroll
    for (int ns = 0; ns < 2; ++ns)
#pragma unroll
      for (int e = 0; e < 8; ++e) {
        int r = rowBase + wr * 64 + ms * 16 + e + 8 * hf;
        int c = colBase + wc * 32 + ns * 16 + ln;
        C[(size_t)r * N + c] = (__bf16)acc[ms][ns][e];
      }
}

// ---------------------------------------------------------------------------
// Kernel 3: TISA bias table. biasTab[h][p], p = (j - i) + 1023, p in [0,2046]
// ---------------------------------------------------------------------------
__global__ void tisa_bias_kernel(const float* __restrict__ amp,
                                 const float* __restrict__ off,
                                 const float* __restrict__ sharp,
                                 float* __restrict__ biasTab) {
  int idx = blockIdx.x * blockDim.x + threadIdx.x;
  if (idx >= 16 * 2047) return;
  int h = idx / 2047, p = idx % 2047;
  float dd = (float)(p - 1023);
  float s = 0.f;
#pragma unroll
  for (int k = 0; k < 21; ++k) {
    float df = dd - off[h * 21 + k];
    s += amp[h * 21 + k] * __expf(-fabsf(sharp[h * 21 + k]) * df * df);
  }
  biasTab[idx] = s;
}

// ---------------------------------------------------------------------------
// Kernel 4: flash attention.  1 WG (128 thr / 4 waves) per (b, h, 64 q rows).
// Each wave owns 16 q rows; KV streamed in 64-wide tiles through LDS.
// K tile staged via TDM DMA; V tile transposed manually.
// ---------------------------------------------------------------------------
__global__ __launch_bounds__(128) void attn_kernel(
    const __bf16* __restrict__ Qb,    // [B*1024][1024]  (col = h*64 + d)
    const __bf16* __restrict__ MemB,  // [B*1024][2048]  (K | V)
    const float* __restrict__ biasTab,
    __bf16* __restrict__ attnOut) {   // [B][1024][16][64]
  __shared__ alignas(16) __bf16 lds_q[64][64];
  __shared__ alignas(16) __bf16 lds_k[64][64];       // [kv][d]
  __shared__ alignas(16) __bf16 lds_vt[64][64];      // [d][kv]
  __shared__ alignas(16) __bf16 lds_p[4][16][64];    // per-wave P scratch

  const int tid  = threadIdx.x;
  const int wave = tid >> 5, lane = tid & 31;
  const int hf   = lane >> 4, ln = lane & 15;
  const int qt = blockIdx.x, h = blockIdx.y, b = blockIdx.z;
  const int qBase = qt * 64;
  const float scale = 0.125f;  // 1/sqrt(64)
  const float* biasH = biasTab + h * 2047;

  // stage Q tile (64 rows x 64 dims): batched loads then stores
  {
    uint4 tq[4];
#pragma unroll
    for (int i = 0; i < 4; ++i) {
      int idx = tid + i * 128;
      int r = idx >> 3, c8 = (idx & 7) * 8;
      tq[i] = *(const uint4*)(Qb + (size_t)(b * 1024 + qBase + r) * 1024 +
                              h * 64 + c8);
    }
#pragma unroll
    for (int i = 0; i < 4; ++i) {
      int idx = tid + i * 128;
      int r = idx >> 3, c8 = (idx & 7) * 8;
      *(uint4*)&lds_q[r][c8] = tq[i];
    }
  }

  float mrow[8], lrow[8];
  v8f o[4] = {};
#pragma unroll
  for (int e = 0; e < 8; ++e) { mrow[e] = -1e30f; lrow[e] = 0.f; }

  for (int t = 0; t < 16; ++t) {
    __syncthreads();  // protect lds_k/lds_vt from previous iteration readers
    const int kvBase = t * 64;
#if ATHENA_HAS_TDM
    if (wave == 0)  // async DMA: K tile 64 rows x 64 dims, row stride 2048
      tdm_load_2d_bf16(MemB + (size_t)(b * 1024 + kvBase) * 2048 + h * 64,
                       &lds_k[0][0], /*tile*/ 64, 64,
                       /*tensor*/ 2048u, 4096u, 2048ull);
    {  // manual V staging with transpose (batched)
      uint4 tv[4];
#pragma unroll
      for (int i = 0; i < 4; ++i) {
        int idx = tid + i * 128;
        int r = idx >> 3, c8 = (idx & 7) * 8;
        tv[i] = *(const uint4*)(
            MemB + (size_t)(b * 1024 + kvBase + r) * 2048 + 1024 + h * 64 + c8);
      }
#pragma unroll
      for (int i = 0; i < 4; ++i) {
        int idx = tid + i * 128;
        int r = idx >> 3, c8 = (idx & 7) * 8;
        const __bf16* s = (const __bf16*)&tv[i];
#pragma unroll
        for (int j = 0; j < 8; ++j) lds_vt[c8 + j][r] = s[j];
      }
    }
    if (wave == 0) __builtin_amdgcn_s_wait_tensorcnt(0);
#else
    {  // manual K + V staging (batched: 8 loads in flight, then stores)
      uint4 tk[4], tv[4];
#pragma unroll
      for (int i = 0; i < 4; ++i) {
        int idx = tid + i * 128;
        int r = idx >> 3, c8 = (idx & 7) * 8;
        const size_t rowOff = (size_t)(b * 1024 + kvBase + r) * 2048 + h * 64;
        tk[i] = *(const uint4*)(MemB + rowOff + c8);
        tv[i] = *(const uint4*)(MemB + rowOff + 1024 + c8);
      }
#pragma unroll
      for (int i = 0; i < 4; ++i) {
        int idx = tid + i * 128;
        int r = idx >> 3, c8 = (idx & 7) * 8;
        *(uint4*)&lds_k[r][c8] = tk[i];
        const __bf16* s = (const __bf16*)&tv[i];
#pragma unroll
        for (int j = 0; j < 8; ++j) lds_vt[c8 + j][r] = s[j];
      }
    }
#endif
    __syncthreads();

    // S = Q @ K^T  (16 q x 64 kv per wave)
    v8f s[4];
#pragma unroll
    for (int nt = 0; nt < 4; ++nt) {
      v8f sacc = {};
#pragma unroll
      for (int kc = 0; kc < 2; ++kc) {
        v16bf a  = load_a_frag(&lds_q[wave * 16 + ln][kc * 32], hf);
        v16bf bb = load_b_frag(&lds_k[nt * 16 + ln][kc * 32 + 16 * hf]);
        sacc = wmma_bf16(a, bb, sacc);
      }
      s[nt] = sacc;
    }

    // scale + bias + tile row-max
    float tmax[8];
#pragma unroll
    for (int e = 0; e < 8; ++e) tmax[e] = -1e30f;
    const int iBase = qBase + wave * 16;
#pragma unroll
    for (int nt = 0; nt < 4; ++nt) {
      int j = kvBase + nt * 16 + ln;
#pragma unroll
      for (int e = 0; e < 8; ++e) {
        int ii = iBase + e + 8 * hf;
        float v = s[nt][e] * scale + biasH[j - ii + 1023];
        s[nt][e] = v;
        tmax[e] = fmaxf(tmax[e], v);
      }
    }
    // row m lives in one 16-lane half -> butterfly over masks 1,2,4,8
#pragma unroll
    for (int e = 0; e < 8; ++e) {
      float v = tmax[e];
      for (int m = 1; m < 16; m <<= 1) v = fmaxf(v, __shfl_xor(v, m, 32));
      tmax[e] = v;
    }
    float alpha[8];
#pragma unroll
    for (int e = 0; e < 8; ++e) {
      float mn = fmaxf(mrow[e], tmax[e]);
      alpha[e] = __expf(mrow[e] - mn);
      mrow[e] = mn;
    }
    // P = exp(S - m), row-sums, spill P to LDS in A-layout-friendly form
    float psum[8];
#pragma unroll
    for (int e = 0; e < 8; ++e) psum[e] = 0.f;
#pragma unroll
    for (int nt = 0; nt < 4; ++nt)
#pragma unroll
      for (int e = 0; e < 8; ++e) {
        float p = __expf(s[nt][e] - mrow[e]);
        psum[e] += p;
        lds_p[wave][e + 8 * hf][nt * 16 + ln] = (__bf16)p;
      }
#pragma unroll
    for (int e = 0; e < 8; ++e) {
      float v = psum[e];
      for (int m = 1; m < 16; m <<= 1) v += __shfl_xor(v, m, 32);
      lrow[e] = lrow[e] * alpha[e] + v;
    }
#pragma unroll
    for (int dt = 0; dt < 4; ++dt)
#pragma unroll
      for (int e = 0; e < 8; ++e) o[dt][e] *= alpha[e];
    __syncthreads();  // make lds_p stores visible to the WMMA reads

    // O += P @ V
#pragma unroll
    for (int dt = 0; dt < 4; ++dt) {
#pragma unroll
      for (int kc = 0; kc < 2; ++kc) {
        v16bf a  = load_a_frag(&lds_p[wave][ln][kc * 32], hf);
        v16bf bb = load_b_frag(&lds_vt[dt * 16 + ln][kc * 32 + 16 * hf]);
        o[dt] = wmma_bf16(a, bb, o[dt]);
      }
    }
  }

  // finalize: O / l, write attn[b][qrow][h][d] (== [B,L,D] head-major)
#pragma unroll
  for (int dt = 0; dt < 4; ++dt)
#pragma unroll
    for (int e = 0; e < 8; ++e) {
      int qrow = qBase + wave * 16 + e + 8 * hf;
      int d = dt * 16 + ln;
      float v = o[dt][e] / lrow[e];
      attnOut[(((size_t)b * 1024 + qrow) * 16 + h) * 64 + d] = (__bf16)v;
    }
}

// ---------------------------------------------------------------------------
// Kernel 5: gated GEMM.  out = (attn@Wg_a + bg_a) * sigmoid(attn@Wg_b + bg_b)
// 256 thr / 8 waves; tile = 128 rows x 64 out-cols (both gate halves).
// ---------------------------------------------------------------------------
__global__ __launch_bounds__(256) void gemm_gated(
    const __bf16* __restrict__ A,    // [4096][1024]
    const __bf16* __restrict__ Wg,   // [1024][2048]
    const float* __restrict__ bg,    // [2048]
    float* __restrict__ out) {       // [4096][1024]
  __shared__ alignas(16) __bf16 lds_a[128][32];
  __shared__ alignas(16) __bf16 lds_bta[64][32];  // a-half, [n][k]
  __shared__ alignas(16) __bf16 lds_btb[64][32];  // b-half, [n][k]

  const int tid  = threadIdx.x;
  const int wave = tid >> 5, lane = tid & 31;
  const int hf   = lane >> 4, ln = lane & 15;
  const int rowBase = blockIdx.y * 128, colBase = blockIdx.x * 64;
  const int wr = wave & 3;   // 32-row group
  const int wc = wave >> 2;  // 32-col group
  const int K = 1024, N = 2048;

  v8f accA[2][2] = {}, accB[2][2] = {};

  for (int k0 = 0; k0 < K; k0 += 32) {
#if ATHENA_HAS_TDM
    if (wave == 0)  // async DMA: A tile 128 x 32
      tdm_load_2d_bf16(A + (size_t)rowBase * K + k0, &lds_a[0][0],
                       /*tile*/ 32, 128, /*tensor*/ 1024u, 4096u, 1024ull);
#else
    {
      uint4 ta[2];
#pragma unroll
      for (int i = 0; i < 2; ++i) {
        int idx = tid + i * 256;
        int r = idx >> 2, c8 = (idx & 3) * 8;
        ta[i] = *(const uint4*)(A + (size_t)(rowBase + r) * K + k0 + c8);
      }
#pragma unroll
      for (int i = 0; i < 2; ++i) {
        int idx = tid + i * 256;
        int r = idx >> 2, c8 = (idx & 3) * 8;
        *(uint4*)&lds_a[r][c8] = ta[i];
      }
    }
#endif
    {  // both gate halves of B: issue both loads, then scatter-transpose
      int kk = tid >> 3, n8 = (tid & 7) * 8;  // 32x64 per half = 256 uint4
      const size_t rowOff = (size_t)(k0 + kk) * N + colBase;
      uint4 da = *(const uint4*)(Wg + rowOff + n8);
      uint4 db = *(const uint4*)(Wg + rowOff + 1024 + n8);
      const __bf16* sa = (const __bf16*)&da;
      const __bf16* sb = (const __bf16*)&db;
#pragma unroll
      for (int j = 0; j < 8; ++j) {
        lds_bta[n8 + j][kk] = sa[j];
        lds_btb[n8 + j][kk] = sb[j];
      }
    }
#if ATHENA_HAS_TDM
    if (wave == 0) __builtin_amdgcn_s_wait_tensorcnt(0);
#endif
    __syncthreads();

    v16bf af[2], bfa[2], bfb[2];
#pragma unroll
    for (int ms = 0; ms < 2; ++ms)
      af[ms] = load_a_frag(&lds_a[wr * 32 + ms * 16 + ln][0], hf);
#pragma unroll
    for (int ns = 0; ns < 2; ++ns) {
      bfa[ns] = load_b_frag(&lds_bta[wc * 32 + ns * 16 + ln][16 * hf]);
      bfb[ns] = load_b_frag(&lds_btb[wc * 32 + ns * 16 + ln][16 * hf]);
    }
#pragma unroll
    for (int ms = 0; ms < 2; ++ms)
#pragma unroll
      for (int ns = 0; ns < 2; ++ns) {
        accA[ms][ns] = wmma_bf16(af[ms], bfa[ns], accA[ms][ns]);
        accB[ms][ns] = wmma_bf16(af[ms], bfb[ns], accB[ms][ns]);
      }
    __syncthreads();
  }
#pragma unroll
  for (int ms = 0; ms < 2; ++ms)
#pragma unroll
    for (int ns = 0; ns < 2; ++ns)
#pragma unroll
      for (int e = 0; e < 8; ++e) {
        int r = rowBase + wr * 32 + ms * 16 + e + 8 * hf;
        int c = colBase + wc * 32 + ns * 16 + ln;
        float ga = accA[ms][ns][e] + bg[c];
        float gb = accB[ms][ns][e] + bg[1024 + c];
        out[(size_t)r * 1024 + c] = ga * (1.f / (1.f + __expf(-gb)));
      }
}

// ---------------------------------------------------------------------------
// Host launcher
// ---------------------------------------------------------------------------
extern "C" void kernel_launch(void* const* d_in, const int* in_sizes, int n_in,
                              void* d_out, int out_size, void* d_ws,
                              size_t ws_size, hipStream_t stream) {
  (void)in_sizes; (void)n_in; (void)out_size; (void)ws_size;

  const float* x_q   = (const float*)d_in[0];  // [4,1024,1024]
  const float* x_kv  = (const float*)d_in[1];  // [4,1024,1024]
  const float* Wq    = (const float*)d_in[2];  // [1024,1024]
  const float* Wm    = (const float*)d_in[3];  // [1024,2048]
  const float* Wg    = (const float*)d_in[4];  // [1024,2048]
  const float* bg    = (const float*)d_in[5];  // [2048]
  const float* amp   = (const float*)d_in[6];  // [16,21]
  const float* off   = (const float*)d_in[7];  // [16,21]
  const float* sharp = (const float*)d_in[8];  // [16,21]
  float* out = (float*)d_out;                  // [4,1024,1024]

  // workspace layout (all offsets 256B-aligned)
  char* ws = (char*)d_ws;
  const size_t N_XQ = 4096ull * 1024, N_WQ = 1024ull * 1024,
               N_WM = 1024ull * 2048, N_WG = 1024ull * 2048;
  __bf16* xq_bf  = (__bf16*)(ws);                        // 8 MB
  __bf16* xkv_bf = (__bf16*)(ws + (8u << 20));           // 8 MB
  __bf16* Wq_bf  = (__bf16*)(ws + (16u << 20));          // 2 MB
  __bf16* Wm_bf  = (__bf16*)(ws + (18u << 20));          // 4 MB
  __bf16* Wg_bf  = (__bf16*)(ws + (22u << 20));          // 4 MB
  __bf16* q_bf   = (__bf16*)(ws + (26u << 20));          // 8 MB
  __bf16* mem_bf = (__bf16*)(ws + (34u << 20));          // 16 MB
  __bf16* att_bf = (__bf16*)(ws + (50u << 20));          // 8 MB
  float*  biasT  = (float*) (ws + (58u << 20));          // 128 KB

  // 0) f32 -> bf16
  cvt_f32_bf16<<<2048, 256, 0, stream>>>(x_q,  xq_bf,  (int)N_XQ);
  cvt_f32_bf16<<<2048, 256, 0, stream>>>(x_kv, xkv_bf, (int)N_XQ);
  cvt_f32_bf16<<<1024, 256, 0, stream>>>(Wq,   Wq_bf,  (int)N_WQ);
  cvt_f32_bf16<<<2048, 256, 0, stream>>>(Wm,   Wm_bf,  (int)N_WM);
  cvt_f32_bf16<<<2048, 256, 0, stream>>>(Wg,   Wg_bf,  (int)N_WG);

  // 1) q = x_q @ Wq   (4096x1024x1024)
  gemm_bf16<<<dim3(1024 / 128, 4096 / 128), 256, 0, stream>>>(
      xq_bf, Wq_bf, q_bf, 4096, 1024, 1024);
  // 2) mem = x_kv @ Wm  (4096x2048x1024) -> K | V
  gemm_bf16<<<dim3(2048 / 128, 4096 / 128), 256, 0, stream>>>(
      xkv_bf, Wm_bf, mem_bf, 4096, 2048, 1024);
  // 3) TISA bias table
  tisa_bias_kernel<<<(16 * 2047 + 255) / 256, 256, 0, stream>>>(amp, off, sharp,
                                                                biasT);
  // 4) flash attention: grid (q-tiles, heads, batch)
  attn_kernel<<<dim3(16, 16, 4), 128, 0, stream>>>(q_bf, mem_bf, biasT, att_bf);
  // 5) gated output GEMM with fused a*sigmoid(b)
  gemm_gated<<<dim3(1024 / 64, 4096 / 128), 256, 0, stream>>>(att_bf, Wg_bf, bg,
                                                              out);
}
